// lwinnn_model_77756087927187
// MI455X (gfx1250) — matrix operand order repaired and checked by reading, not compiled, not run on hardware
//
#include <hip/hip_runtime.h>
#include <cmath>

typedef __attribute__((ext_vector_type(16))) __bf16 v16bf;
typedef __attribute__((ext_vector_type(8)))  float  v8f;

#define H1 56
#define W1 56
#define BB 16
#define NN 32
#define CC 512
#define WIN 5
#define HM 60   // H1 + WIN - 1
#define WM 60
#define OH 224
#define OW 224
#define KS 33
#define KH 16   // KS/2

struct GK { float k[KS]; };

__device__ __forceinline__ float sq4(float4 f) {
    return f.x*f.x + f.y*f.y + f.z*f.z + f.w*f.w;
}

template<int I>
__device__ __forceinline__ void pk4(v16bf& v, float4 f) {
    v[I+0] = (__bf16)f.x; v[I+1] = (__bf16)f.y;
    v[I+2] = (__bf16)f.z; v[I+3] = (__bf16)f.w;
}

// ---------------------------------------------------------------------------
// Kernel 1: convert memory_bank fp32 -> bf16, compute per-row sum of squares.
// One wave (32 lanes) per row of 512 elements. rows = 60*60*32 = 115200.
// ---------------------------------------------------------------------------
__global__ void cvt_mem_kernel(const float* __restrict__ mb,
                               unsigned short* __restrict__ mbB_u,
                               float* __restrict__ ynorm) {
    int wid  = (int)((blockIdx.x * blockDim.x + threadIdx.x) >> 5);
    int lane = (int)(threadIdx.x & 31u);
    if (wid >= HM * WM * NN) return;
    const float* src = mb + (size_t)wid * CC + lane * 16;
    float4 f0 = ((const float4*)src)[0];
    float4 f1 = ((const float4*)src)[1];
    float4 f2 = ((const float4*)src)[2];
    float4 f3 = ((const float4*)src)[3];
    float sq = sq4(f0) + sq4(f1) + sq4(f2) + sq4(f3);
    v16bf v;
    pk4<0>(v, f0); pk4<4>(v, f1); pk4<8>(v, f2); pk4<12>(v, f3);
    __bf16* dst = (__bf16*)mbB_u + (size_t)wid * CC + lane * 16;
    *(v16bf*)dst = v;
    #pragma unroll
    for (int m = 16; m >= 1; m >>= 1) sq += __shfl_xor(sq, m, 32);
    if (lane == 0) ynorm[wid] = sq;
}

// ---------------------------------------------------------------------------
// Kernel 2: main distance kernel. One wave per (h,w) position.
//   cross = A(16x512) @ B(512x32) via v_wmma_f32_16x16x32_bf16 (2 N-tiles),
//   dsq = max(|x|^2 - 2*cross + |y|^2, 0); min over n; min over 25 offsets.
// ---------------------------------------------------------------------------
__global__ void __launch_bounds__(128, 1)
dist_kernel(const float* __restrict__ emb,
            const unsigned short* __restrict__ mbB_u,
            const float* __restrict__ ynorm,
            float* __restrict__ pix) {
    const __bf16* mbB = (const __bf16*)mbB_u;
    int wave = (int)((blockIdx.x * blockDim.x + threadIdx.x) >> 5);
    int lane = (int)(threadIdx.x & 31u);
    int half = lane >> 4;       // 0: lanes 0-15, 1: lanes 16-31
    int lm   = lane & 15;
    int h = wave / W1, w = wave % W1;

    // --- Build A fragments in registers + x-norm partial sums ---
    // A k order used here (contiguous, matches B): lane lm (half 0) holds K c*32+0..15
    //                           lane lm (half 1) holds K c*32+16..31
    v16bf a[16];
    float sq = 0.f;
    const float* arow = emb + ((size_t)(h * W1 + w) * BB + lm) * CC;
    #pragma unroll
    for (int c = 0; c < 16; ++c) {
        int base = c * 32 + half * 16;
        float4 f0 = *(const float4*)(arow + base);
        float4 f1 = *(const float4*)(arow + base + 4);
        float4 f2 = *(const float4*)(arow + base + 8);
        float4 f3 = *(const float4*)(arow + base + 12);
        sq += sq4(f0) + sq4(f1) + sq4(f2) + sq4(f3);
        v16bf v;
        pk4<0>(v, f0); pk4<4>(v, f1); pk4<8>(v, f2); pk4<12>(v, f3);
        a[c] = v;
    }
    // lane L and L+16 together cover row lm's full K=512
    sq += __shfl_xor(sq, 16, 32);           // lane m (and m+16) holds xnorm[m]
    float xnr[8];
    #pragma unroll
    for (int r = 0; r < 8; ++r) xnr[r] = __shfl(sq, half ? (8 + r) : r, 32);

    float best[8];
    #pragma unroll
    for (int r = 0; r < 8; ++r) best[r] = 1e30f;

    for (int d1 = 0; d1 < WIN; ++d1) {
        for (int d2 = 0; d2 < WIN; ++d2) {
            int ypos = (h + d1) * WM + (w + d2);
            // B k order used here (contiguous, matches A): lane n (half 0) holds K 0..15 of col n,
            // lane n (half 1) holds K 16..31 -> contiguous along C in memory.
            const __bf16* brow0 = mbB + ((size_t)ypos * NN + lm) * CC + half * 16;
            const __bf16* brow1 = brow0 + (size_t)16 * CC;   // cols n=16..31
            v8f acc0 = {0.f,0.f,0.f,0.f,0.f,0.f,0.f,0.f};
            v8f acc1 = {0.f,0.f,0.f,0.f,0.f,0.f,0.f,0.f};
            #pragma unroll
            for (int c = 0; c < 16; ++c) {
                v16bf b0 = *(const v16bf*)(brow0 + c * 32);
                v16bf b1 = *(const v16bf*)(brow1 + c * 32);
                acc0 = __builtin_amdgcn_wmma_f32_16x16x32_bf16(
                    false, a[c], false, b0, (short)0, acc0, false, false);
                acc1 = __builtin_amdgcn_wmma_f32_16x16x32_bf16(
                    false, a[c], false, b1, (short)0, acc1, false, false);
            }
            float yn0 = ynorm[ypos * NN + lm];
            float yn1 = ynorm[ypos * NN + 16 + lm];
            // C-layout: lane (N = lane%16), VGPR r -> M = r + 8*half
            #pragma unroll
            for (int r = 0; r < 8; ++r) {
                float d0 = fmaxf(xnr[r] - 2.f * acc0[r] + yn0, 0.f);
                float d1v = fmaxf(xnr[r] - 2.f * acc1[r] + yn1, 0.f);
                float dm = fminf(d0, d1v);                 // min over n-tiles
                dm = fminf(dm, __shfl_xor(dm, 1, 32));     // min over n within half
                dm = fminf(dm, __shfl_xor(dm, 2, 32));
                dm = fminf(dm, __shfl_xor(dm, 4, 32));
                dm = fminf(dm, __shfl_xor(dm, 8, 32));
                best[r] = fminf(best[r], dm);              // min over offsets
            }
        }
    }
    int posIdx = h * W1 + w;
    #pragma unroll
    for (int r = 0; r < 8; ++r) {
        if (lm == r)
            pix[(size_t)(r + 8 * half) * (H1 * W1) + posIdx] = sqrtf(best[r]);
    }
}

// ---------------------------------------------------------------------------
// Kernel 3: per-batch max over 56x56 -> d_out[0..15]
// ---------------------------------------------------------------------------
__global__ void img_max_kernel(const float* __restrict__ pix,
                               float* __restrict__ out) {
    __shared__ float red[256];
    int b = blockIdx.x;
    float m = 0.f;  // pix values are sqrt() >= 0
    for (int i = (int)threadIdx.x; i < H1 * W1; i += 256)
        m = fmaxf(m, pix[b * H1 * W1 + i]);
    red[threadIdx.x] = m;
    __syncthreads();
    for (int s = 128; s > 0; s >>= 1) {
        if ((int)threadIdx.x < s)
            red[threadIdx.x] = fmaxf(red[threadIdx.x], red[threadIdx.x + s]);
        __syncthreads();
    }
    if (threadIdx.x == 0) out[b] = red[0];
}

// ---------------------------------------------------------------------------
// Kernel 4: bilinear resize 56x56 -> 224x224 (half-pixel centers, edge clamp)
// ---------------------------------------------------------------------------
__global__ void resize_kernel(const float* __restrict__ pix,
                              float* __restrict__ res) {
    int idx = (int)(blockIdx.x * blockDim.x + threadIdx.x);
    if (idx >= BB * OH * OW) return;
    int x = idx % OW;
    int y = (idx / OW) % OH;
    int b = idx / (OH * OW);
    float cy = (y + 0.5f) * 0.25f - 0.5f;
    float cx = (x + 0.5f) * 0.25f - 0.5f;
    int y0 = (int)floorf(cy); float fy = cy - (float)y0;
    int x0 = (int)floorf(cx); float fx = cx - (float)x0;
    int ya = min(max(y0, 0), H1 - 1), yb = min(max(y0 + 1, 0), H1 - 1);
    int xa = min(max(x0, 0), W1 - 1), xb = min(max(x0 + 1, 0), W1 - 1);
    const float* p = pix + b * H1 * W1;
    float v = (p[ya * W1 + xa] * (1.f - fx) + p[ya * W1 + xb] * fx) * (1.f - fy)
            + (p[yb * W1 + xa] * (1.f - fx) + p[yb * W1 + xb] * fx) * fy;
    res[idx] = v;
}

// ---------------------------------------------------------------------------
// Kernels 5/6: separable 33-tap Gaussian with reflect padding
// ---------------------------------------------------------------------------
__global__ void blur_v_kernel(const float* __restrict__ res,
                              float* __restrict__ tmp, GK gk) {
    int idx = (int)(blockIdx.x * blockDim.x + threadIdx.x);
    if (idx >= BB * OH * OW) return;
    int x = idx % OW;
    int y = (idx / OW) % OH;
    int b = idx / (OH * OW);
    float s = 0.f;
    #pragma unroll
    for (int t = 0; t < KS; ++t) {
        int sy = y + t - KH;
        sy = (sy < 0) ? -sy : ((sy > OH - 1) ? 2 * (OH - 1) - sy : sy);
        s += gk.k[t] * res[((size_t)b * OH + sy) * OW + x];
    }
    tmp[idx] = s;
}

__global__ void blur_h_kernel(const float* __restrict__ tmp,
                              float* __restrict__ out, GK gk) {
    int idx = (int)(blockIdx.x * blockDim.x + threadIdx.x);
    if (idx >= BB * OH * OW) return;
    int x = idx % OW;
    int y = (idx / OW) % OH;
    int b = idx / (OH * OW);
    float s = 0.f;
    #pragma unroll
    for (int t = 0; t < KS; ++t) {
        int sx = x + t - KH;
        sx = (sx < 0) ? -sx : ((sx > OW - 1) ? 2 * (OW - 1) - sx : sx);
        s += gk.k[t] * tmp[((size_t)b * OH + y) * OW + sx];
    }
    out[idx] = s;
}

// ---------------------------------------------------------------------------
extern "C" void kernel_launch(void* const* d_in, const int* in_sizes, int n_in,
                              void* d_out, int out_size, void* d_ws, size_t ws_size,
                              hipStream_t stream) {
    const float* emb = (const float*)d_in[0];   // (56,56,16,512) fp32
    const float* mb  = (const float*)d_in[1];   // (60,60,32,512) fp32
    float* out = (float*)d_out;                 // [16 img] + [16*224*224 blur]

    // Workspace layout (all sizes are multiples of 256 bytes)
    char* ws = (char*)d_ws;
    size_t off = 0;
    unsigned short* mbB = (unsigned short*)(ws + off); off += (size_t)HM * WM * NN * CC * 2;
    float* ynorm = (float*)(ws + off);                 off += (size_t)HM * WM * NN * 4;
    float* pix   = (float*)(ws + off);                 off += (size_t)BB * H1 * W1 * 4;
    float* res   = (float*)(ws + off);                 off += (size_t)BB * OH * OW * 4;
    float* tmp   = (float*)(ws + off);                 off += (size_t)BB * OH * OW * 4;
    (void)ws_size; (void)in_sizes; (void)n_in; (void)out_size;

    // Host-side Gaussian coefficients (sigma=4, 33 taps, normalized)
    GK gk;
    {
        float sum = 0.f;
        for (int i = 0; i < KS; ++i) {
            float x = (float)i - (float)KH;
            gk.k[i] = expf(-0.5f * (x / 4.0f) * (x / 4.0f));
            sum += gk.k[i];
        }
        for (int i = 0; i < KS; ++i) gk.k[i] /= sum;
    }

    // 1) convert memory_bank + y-norms: 115200 rows, 8 waves/block
    cvt_mem_kernel<<<(HM * WM * NN) / 8, 256, 0, stream>>>(mb, mbB, ynorm);

    // 2) main WMMA distance kernel: 3136 waves, 4 waves/block
    dist_kernel<<<(H1 * W1) / 4, 128, 0, stream>>>(emb, mbB, ynorm, pix);

    // 3) per-batch max -> out[0..15]
    img_max_kernel<<<BB, 256, 0, stream>>>(pix, out);

    // 4) bilinear resize 56 -> 224
    int npix = BB * OH * OW;
    resize_kernel<<<(npix + 255) / 256, 256, 0, stream>>>(pix, res);

    // 5/6) separable Gaussian blur (vertical then horizontal) -> out[16..]
    blur_v_kernel<<<(npix + 255) / 256, 256, 0, stream>>>(res, tmp, gk);
    blur_h_kernel<<<(npix + 255) / 256, 256, 0, stream>>>(tmp, out + BB, gk);
}
